// BailingMoEBlock_28063316312109
// MI455X (gfx1250) — compile-verified, matrix operand order checked by reading, not run on hardware
//
#include <hip/hip_runtime.h>
#include <hip/hip_bf16.h>

// ---------------- problem constants (from reference) ----------------
#define T_TOK   1024
#define HID     1024
#define NEXP    16
#define TOPK    4
#define MOE_I   512
#define SH_I    1024          // SHARED_INTER
#define SH_GU   (2 * SH_I)    // merged gate_up width = 2048

#define KPAD    40            // LDS tile row stride in bf16 (32 data + 8 pad; 80B, 16B-aligned)

typedef __attribute__((ext_vector_type(16))) __bf16 v16bf;
typedef __attribute__((ext_vector_type(8)))  float  v8f;

// ---- CDNA5 async global->LDS staging (ASYNCcnt path, ISA 15.18.3 op 98) ----
// LDS dest VGPR = workgroup-relative LDS byte offset (= low 32 bits of a
// generic pointer to __shared__). INST_OFFSET is added to BOTH addresses.
#define ASYNC_LDS_B128(lds_off, gsrc)                                          \
    asm volatile("global_load_async_to_lds_b128 %0, %1, off"                   \
                 :: "v"(lds_off), "v"(gsrc) : "memory")
#define ASYNC_LDS_B128_O16(lds_off, gsrc)                                      \
    asm volatile("global_load_async_to_lds_b128 %0, %1, off offset:16"         \
                 :: "v"(lds_off), "v"(gsrc) : "memory")
#define WAIT_ASYNC0()                                                          \
    asm volatile("s_wait_asynccnt 0x0" ::: "memory")

__device__ __forceinline__ uint32_t lds_off32(const void* p) {
    return (uint32_t)(uintptr_t)p;
}

// ---------------- WMMA fragment helper (wave32, 16x16x32 bf16) ----------------
// A (16x32) and B (staged [n][k], K contiguous) share the layout:
//   row = lane&15; VGPR v holds K pair at k0 = (v&3)*2 + (lane>=16?8:0) + (v>=4?16:0)
// -> two contiguous 16B chunks per lane => 2x ds_load_b128.
// C/D (16x16 f32): VGPR r -> M = r + 8*(lane>=16), N = lane&15.
__device__ __forceinline__ v16bf load_frag(const __bf16* rowp, int hlf) {
    v16bf a;
#pragma unroll
    for (int v = 0; v < 8; ++v) {
        const int k0 = ((v & 3) << 1) + (hlf << 3) + ((v >> 2) << 4);
        a[2 * v]     = rowp[k0];
        a[2 * v + 1] = rowp[k0 + 1];
    }
    return a;
}

__device__ __forceinline__ v8f wmma_bf16(const v16bf& a, const v16bf& b, const v8f& c) {
    return __builtin_amdgcn_wmma_f32_16x16x32_bf16(
        false, a, false, b, (short)0, c, false, false);
}

__device__ __forceinline__ float silu_f(float x) {
    return x / (1.0f + __expf(-x));
}

// ---------------- kernel 1a: plain f32 -> bf16 ----------------
__global__ void cvt_f32_bf16(const float* __restrict__ s, __bf16* __restrict__ d, int n) {
    int i = blockIdx.x * blockDim.x + threadIdx.x;
    const int stride = gridDim.x * blockDim.x;
    for (; i < n; i += stride) d[i] = (__bf16)s[i];
}

// ---------------- kernel 1b: tiled transpose + f32 -> bf16 ----------------
// src: f32 [R][C] (batched), dst: bf16 [C][R].  grid = (C/32, R/32, batch), block 256.
__global__ void cvt_transpose_bf16(const float* __restrict__ src, __bf16* __restrict__ dst,
                                   int R, int C) {
    const size_t bo = (size_t)blockIdx.z * R * C;
    const float* s = src + bo;
    __bf16* d = dst + bo;
    const int c0 = blockIdx.x * 32, r0 = blockIdx.y * 32;
    __shared__ float t[32][33];
    const int cr = threadIdx.x & 31, rr = threadIdx.x >> 5;  // rr in 0..7
#pragma unroll
    for (int l = 0; l < 4; ++l)
        t[rr + l * 8][cr] = s[(size_t)(r0 + rr + l * 8) * C + c0 + cr];
    __syncthreads();
#pragma unroll
    for (int l = 0; l < 4; ++l)
        d[(size_t)(c0 + rr + l * 8) * R + r0 + cr] = (__bf16)t[cr][rr + l * 8];
}

// ---------------- kernel 2: router (softmax + top-4, normalized) ----------------
__global__ void router_kernel(const float* __restrict__ x,
                              const float* __restrict__ rw,
                              float* __restrict__ combine) {
    const int t = blockIdx.x * blockDim.x + threadIdx.x;
    if (t >= T_TOK) return;
    float logits[NEXP];
#pragma unroll
    for (int e = 0; e < NEXP; ++e) {
        float s = 0.f;
        const float* xr = x + (size_t)t * HID;
        const float* wr = rw + (size_t)e * HID;
        for (int h = 0; h < HID; ++h) s += xr[h] * wr[h];
        logits[e] = s;
    }
    float mx = logits[0];
#pragma unroll
    for (int e = 1; e < NEXP; ++e) mx = fmaxf(mx, logits[e]);
    float probs[NEXP];
    float den = 0.f;
#pragma unroll
    for (int e = 0; e < NEXP; ++e) { probs[e] = __expf(logits[e] - mx); den += probs[e]; }
#pragma unroll
    for (int e = 0; e < NEXP; ++e) probs[e] /= den;
    float cw[NEXP];
#pragma unroll
    for (int e = 0; e < NEXP; ++e) cw[e] = 0.f;
    float wsum = 0.f;
#pragma unroll
    for (int k = 0; k < TOPK; ++k) {
        int bi = 0; float bv = -1.f;
#pragma unroll
        for (int e = 0; e < NEXP; ++e)
            if (probs[e] > bv) { bv = probs[e]; bi = e; }
        cw[bi] = bv; wsum += bv; probs[bi] = -2.f;
    }
    const float inv = 1.0f / wsum;  // NORM_TOPK, ROUTED_SCALE = 1
#pragma unroll
    for (int e = 0; e < NEXP; ++e) combine[t * NEXP + e] = cw[e] * inv;
}

// ---------------- kernel 3: fused gate/up GEMM + SwiGLU (double-buffered async) ----
// Weights pre-transposed: Wg_t/Wu_t are [Ncols][Hk] row-major (K contiguous).
// Block 256 (8 waves): tile 128(M) x 64(N); waves 4x2, wave tile 32x32 = 2x2 frags x {G,U}.
// Double-buffer loop unrolled by 2 so buffer indices are compile-time constants
// (keeps addrspace(3) inference -> ds_load_b128 fragment loads).
// Requires Hk % 64 == 0.  grid = (Ncols/64, T/128, E)
__global__ void gateup_kernel(const __bf16* __restrict__ xb,   // [T][Hk]
                              const __bf16* __restrict__ wgt,  // gate^T base [N][Hk]
                              const __bf16* __restrict__ wut,  // up^T base   [N][Hk]
                              long wstride,                    // per-expert weight stride
                              __bf16* __restrict__ hout,       // activated output
                              long hstride, int ldh, int Hk) {
    const int e = blockIdx.z;
    const __bf16* wgE = wgt + (long)e * wstride;
    const __bf16* wuE = wut + (long)e * wstride;
    __bf16* hE = hout + (long)e * hstride;
    const int n0g = blockIdx.x * 64;
    const int t0g = blockIdx.y * 128;

    __shared__ __bf16 At[2][128 * KPAD];
    __shared__ __bf16 Bg[2][64 * KPAD];
    __shared__ __bf16 Bu[2][64 * KPAD];

    const int tid = threadIdx.x;
    const int wave = tid >> 5, lane = tid & 31;
    const int m0 = (wave >> 1) * 32;
    const int n0 = (wave & 1) * 32;
    const int hlf = lane >> 4, lr = lane & 15;

    // staging map: A = 128 rows x 16 u32 (8 u32/thread); B = 64 rows x 16 u32 (4 u32/thread)
    const int a_row = tid >> 1, a_cu = (tid & 1) * 8;
    const int b_row = tid >> 2, b_cu = (tid & 3) * 4;
    const uint32_t* a_src  = (const uint32_t*)(xb  + (size_t)(t0g + a_row) * Hk) + a_cu;
    const uint32_t* bg_src = (const uint32_t*)(wgE + (size_t)(n0g + b_row) * Hk) + b_cu;
    const uint32_t* bu_src = (const uint32_t*)(wuE + (size_t)(n0g + b_row) * Hk) + b_cu;

    uint32_t aL[2], gL[2], uL[2];
#pragma unroll
    for (int b = 0; b < 2; ++b) {
        aL[b] = lds_off32((uint32_t*)&At[b][0] + a_row * (KPAD / 2) + a_cu);
        gL[b] = lds_off32((uint32_t*)&Bg[b][0] + b_row * (KPAD / 2) + b_cu);
        uL[b] = lds_off32((uint32_t*)&Bu[b][0] + b_row * (KPAD / 2) + b_cu);
    }

    // loop-invariant fragment row pointers (only ever indexed by literals)
    const __bf16 *apr[2][2], *bgr[2][2], *bur[2][2];
#pragma unroll
    for (int b = 0; b < 2; ++b) {
#pragma unroll
        for (int i = 0; i < 2; ++i) apr[b][i] = &At[b][0] + (m0 + i * 16 + lr) * KPAD;
#pragma unroll
        for (int j = 0; j < 2; ++j) {
            bgr[b][j] = &Bg[b][0] + (n0 + j * 16 + lr) * KPAD;
            bur[b][j] = &Bu[b][0] + (n0 + j * 16 + lr) * KPAD;
        }
    }

    v8f zero = {0.f, 0.f, 0.f, 0.f, 0.f, 0.f, 0.f, 0.f};
    v8f accG[2][2], accU[2][2];
#pragma unroll
    for (int i = 0; i < 2; ++i)
#pragma unroll
        for (int j = 0; j < 2; ++j) { accG[i][j] = zero; accU[i][j] = zero; }

#define GU_STAGE(B) do {                                                       \
        ASYNC_LDS_B128(aL[B], a_src);                                          \
        ASYNC_LDS_B128_O16(aL[B], a_src);                                      \
        ASYNC_LDS_B128(gL[B], bg_src);                                         \
        ASYNC_LDS_B128(uL[B], bu_src);                                         \
        a_src += 16; bg_src += 16; bu_src += 16;                               \
    } while (0)

#define GU_COMPUTE(B) do {                                                     \
        const v16bf a0  = load_frag(apr[B][0], hlf);                           \
        const v16bf a1  = load_frag(apr[B][1], hlf);                           \
        const v16bf bg0 = load_frag(bgr[B][0], hlf);                           \
        const v16bf bu0 = load_frag(bur[B][0], hlf);                           \
        accG[0][0] = wmma_bf16(a0, bg0, accG[0][0]);                           \
        accG[1][0] = wmma_bf16(a1, bg0, accG[1][0]);                           \
        accU[0][0] = wmma_bf16(a0, bu0, accU[0][0]);                           \
        accU[1][0] = wmma_bf16(a1, bu0, accU[1][0]);                           \
        const v16bf bg1 = load_frag(bgr[B][1], hlf);                           \
        const v16bf bu1 = load_frag(bur[B][1], hlf);                           \
        accG[0][1] = wmma_bf16(a0, bg1, accG[0][1]);                           \
        accG[1][1] = wmma_bf16(a1, bg1, accG[1][1]);                           \
        accU[0][1] = wmma_bf16(a0, bu1, accU[0][1]);                           \
        accU[1][1] = wmma_bf16(a1, bu1, accU[1][1]);                           \
    } while (0)

    const int nsteps = Hk >> 5;   // even (Hk % 64 == 0)
    GU_STAGE(0);
    WAIT_ASYNC0();
    __syncthreads();
    for (int s = 0; s < nsteps; s += 2) {
        if (s + 1 < nsteps) GU_STAGE(1);
        GU_COMPUTE(0);
        WAIT_ASYNC0();
        __syncthreads();
        if (s + 2 < nsteps) GU_STAGE(0);
        GU_COMPUTE(1);
        WAIT_ASYNC0();
        __syncthreads();
    }
#undef GU_STAGE
#undef GU_COMPUTE

    // SwiGLU + bf16 store
#pragma unroll
    for (int i = 0; i < 2; ++i)
#pragma unroll
        for (int j = 0; j < 2; ++j)
#pragma unroll
            for (int r = 0; r < 8; ++r) {
                const float g = accG[i][j][r];
                const float u = accU[i][j][r];
                const int row = t0g + m0 + i * 16 + r + hlf * 8;
                const int col = n0g + n0 + j * 16 + lr;
                hE[(size_t)row * ldh + col] = (__bf16)(silu_f(g) * u);
            }
}

// ---------------- kernel 4: routed down-proj with combine weights ----------------
// wdt: [E][HID][MOE_I] (w_down^T per expert; K = MOE_I contiguous)
// grid = (HID/64, T/128); loops 16 experts; accP += combine[t,e] * tileE; writes out.
__global__ void down_routed_kernel(const __bf16* __restrict__ hb,    // [E][T][MOE_I]
                                   const __bf16* __restrict__ wdt,   // [E][HID][MOE_I]
                                   const float* __restrict__ combine,// [T][E]
                                   float* __restrict__ out) {        // [T][HID]
    const int n0g = blockIdx.x * 64;
    const int t0g = blockIdx.y * 128;

    __shared__ __bf16 At[2][128 * KPAD];
    __shared__ __bf16 Bt[2][64 * KPAD];
    __shared__ float  comb[128 * NEXP];

    const int tid = threadIdx.x;
    const int wave = tid >> 5, lane = tid & 31;
    const int m0 = (wave >> 1) * 32;
    const int n0 = (wave & 1) * 32;
    const int hlf = lane >> 4, lr = lane & 15;

#pragma unroll
    for (int j = 0; j < 8; ++j) {
        const int u = tid * 8 + j;
        comb[u] = combine[(size_t)(t0g + (u >> 4)) * NEXP + (u & 15)];
    }

    const int a_row = tid >> 1, a_cu = (tid & 1) * 8;
    const int b_row = tid >> 2, b_cu = (tid & 3) * 4;
    uint32_t aL[2], bL[2];
#pragma unroll
    for (int b = 0; b < 2; ++b) {
        aL[b] = lds_off32((uint32_t*)&At[b][0] + a_row * (KPAD / 2) + a_cu);
        bL[b] = lds_off32((uint32_t*)&Bt[b][0] + b_row * (KPAD / 2) + b_cu);
    }

    const __bf16 *apr[2][2], *bpr[2][2];
#pragma unroll
    for (int b = 0; b < 2; ++b) {
#pragma unroll
        for (int i = 0; i < 2; ++i) apr[b][i] = &At[b][0] + (m0 + i * 16 + lr) * KPAD;
#pragma unroll
        for (int j = 0; j < 2; ++j) bpr[b][j] = &Bt[b][0] + (n0 + j * 16 + lr) * KPAD;
    }

    v8f zero = {0.f, 0.f, 0.f, 0.f, 0.f, 0.f, 0.f, 0.f};
    v8f accP[2][2];
#pragma unroll
    for (int i = 0; i < 2; ++i)
#pragma unroll
        for (int j = 0; j < 2; ++j) accP[i][j] = zero;

    for (int e = 0; e < NEXP; ++e) {
        const uint32_t* a_src =
            (const uint32_t*)(hb  + (size_t)e * T_TOK * MOE_I + (size_t)(t0g + a_row) * MOE_I) + a_cu;
        const uint32_t* b_src =
            (const uint32_t*)(wdt + (size_t)e * MOE_I * HID   + (size_t)(n0g + b_row) * MOE_I) + b_cu;
        v8f accT[2][2];
#pragma unroll
        for (int i = 0; i < 2; ++i)
#pragma unroll
            for (int j = 0; j < 2; ++j) accT[i][j] = zero;

#define DR_STAGE(B) do {                                                       \
            ASYNC_LDS_B128(aL[B], a_src);                                      \
            ASYNC_LDS_B128_O16(aL[B], a_src);                                  \
            ASYNC_LDS_B128(bL[B], b_src);                                      \
            a_src += 16; b_src += 16;                                          \
        } while (0)

#define DR_COMPUTE(B) do {                                                     \
            const v16bf a0 = load_frag(apr[B][0], hlf);                        \
            const v16bf a1 = load_frag(apr[B][1], hlf);                        \
            const v16bf b0 = load_frag(bpr[B][0], hlf);                        \
            accT[0][0] = wmma_bf16(a0, b0, accT[0][0]);                        \
            accT[1][0] = wmma_bf16(a1, b0, accT[1][0]);                        \
            const v16bf b1 = load_frag(bpr[B][1], hlf);                        \
            accT[0][1] = wmma_bf16(a0, b1, accT[0][1]);                        \
            accT[1][1] = wmma_bf16(a1, b1, accT[1][1]);                        \
        } while (0)

        const int nsteps = MOE_I >> 5;  // 16 (even)
        DR_STAGE(0);
        WAIT_ASYNC0();
        __syncthreads();
        for (int s = 0; s < nsteps; s += 2) {
            if (s + 1 < nsteps) DR_STAGE(1);
            DR_COMPUTE(0);
            WAIT_ASYNC0();
            __syncthreads();
            if (s + 2 < nsteps) DR_STAGE(0);
            DR_COMPUTE(1);
            WAIT_ASYNC0();
            __syncthreads();
        }
#undef DR_STAGE
#undef DR_COMPUTE

        // fold: accP += combine[t,e] * accT
#pragma unroll
        for (int i = 0; i < 2; ++i)
#pragma unroll
            for (int r = 0; r < 8; ++r) {
                const int ml = m0 + i * 16 + r + hlf * 8;
                const float sc = comb[ml * NEXP + e];
#pragma unroll
                for (int j = 0; j < 2; ++j)
                    accP[i][j][r] += sc * accT[i][j][r];
            }
    }

#pragma unroll
    for (int i = 0; i < 2; ++i)
#pragma unroll
        for (int j = 0; j < 2; ++j)
#pragma unroll
            for (int r = 0; r < 8; ++r) {
                const int row = t0g + m0 + i * 16 + r + hlf * 8;
                const int col = n0g + n0 + j * 16 + lr;
                out[(size_t)row * HID + col] = accP[i][j][r];
            }
}

// ---------------- kernel 5: shared-expert down-proj (out +=) ----------------
// wsdt: [HID][SH_I] (ws_down^T; K = SH_I contiguous)
__global__ void down_shared_kernel(const __bf16* __restrict__ hsb,   // [T][SH_I]
                                   const __bf16* __restrict__ wsdt,  // [HID][SH_I]
                                   float* __restrict__ out) {
    const int n0g = blockIdx.x * 64;
    const int t0g = blockIdx.y * 128;

    __shared__ __bf16 At[2][128 * KPAD];
    __shared__ __bf16 Bt[2][64 * KPAD];

    const int tid = threadIdx.x;
    const int wave = tid >> 5, lane = tid & 31;
    const int m0 = (wave >> 1) * 32;
    const int n0 = (wave & 1) * 32;
    const int hlf = lane >> 4, lr = lane & 15;

    const int a_row = tid >> 1, a_cu = (tid & 1) * 8;
    const int b_row = tid >> 2, b_cu = (tid & 3) * 4;
    const uint32_t* a_src = (const uint32_t*)(hsb  + (size_t)(t0g + a_row) * SH_I) + a_cu;
    const uint32_t* b_src = (const uint32_t*)(wsdt + (size_t)(n0g + b_row) * SH_I) + b_cu;
    uint32_t aL[2], bL[2];
#pragma unroll
    for (int b = 0; b < 2; ++b) {
        aL[b] = lds_off32((uint32_t*)&At[b][0] + a_row * (KPAD / 2) + a_cu);
        bL[b] = lds_off32((uint32_t*)&Bt[b][0] + b_row * (KPAD / 2) + b_cu);
    }

    const __bf16 *apr[2][2], *bpr[2][2];
#pragma unroll
    for (int b = 0; b < 2; ++b) {
#pragma unroll
        for (int i = 0; i < 2; ++i) apr[b][i] = &At[b][0] + (m0 + i * 16 + lr) * KPAD;
#pragma unroll
        for (int j = 0; j < 2; ++j) bpr[b][j] = &Bt[b][0] + (n0 + j * 16 + lr) * KPAD;
    }

    v8f zero = {0.f, 0.f, 0.f, 0.f, 0.f, 0.f, 0.f, 0.f};
    v8f acc[2][2];
#pragma unroll
    for (int i = 0; i < 2; ++i)
#pragma unroll
        for (int j = 0; j < 2; ++j) acc[i][j] = zero;

#define DS_STAGE(B) do {                                                       \
        ASYNC_LDS_B128(aL[B], a_src);                                          \
        ASYNC_LDS_B128_O16(aL[B], a_src);                                      \
        ASYNC_LDS_B128(bL[B], b_src);                                          \
        a_src += 16; b_src += 16;                                              \
    } while (0)

#define DS_COMPUTE(B) do {                                                     \
        const v16bf a0 = load_frag(apr[B][0], hlf);                            \
        const v16bf a1 = load_frag(apr[B][1], hlf);                            \
        const v16bf b0 = load_frag(bpr[B][0], hlf);                            \
        acc[0][0] = wmma_bf16(a0, b0, acc[0][0]);                              \
        acc[1][0] = wmma_bf16(a1, b0, acc[1][0]);                              \
        const v16bf b1 = load_frag(bpr[B][1], hlf);                            \
        acc[0][1] = wmma_bf16(a0, b1, acc[0][1]);                              \
        acc[1][1] = wmma_bf16(a1, b1, acc[1][1]);                              \
    } while (0)

    const int nsteps = SH_I >> 5;  // 32 (even)
    DS_STAGE(0);
    WAIT_ASYNC0();
    __syncthreads();
    for (int s = 0; s < nsteps; s += 2) {
        if (s + 1 < nsteps) DS_STAGE(1);
        DS_COMPUTE(0);
        WAIT_ASYNC0();
        __syncthreads();
        if (s + 2 < nsteps) DS_STAGE(0);
        DS_COMPUTE(1);
        WAIT_ASYNC0();
        __syncthreads();
    }
#undef DS_STAGE
#undef DS_COMPUTE

#pragma unroll
    for (int i = 0; i < 2; ++i)
#pragma unroll
        for (int j = 0; j < 2; ++j)
#pragma unroll
            for (int r = 0; r < 8; ++r) {
                const int row = t0g + m0 + i * 16 + r + hlf * 8;
                const int col = n0g + n0 + j * 16 + lr;
                out[(size_t)row * HID + col] += acc[i][j][r];  // routed wrote first
            }
}

// ---------------- host launcher ----------------
extern "C" void kernel_launch(void* const* d_in, const int* in_sizes, int n_in,
                              void* d_out, int out_size, void* d_ws, size_t ws_size,
                              hipStream_t stream) {
    (void)in_sizes; (void)n_in; (void)out_size; (void)ws_size;

    const float* x    = (const float*)d_in[0];  // [T, H]
    const float* rw   = (const float*)d_in[1];  // [E, H]
    const float* wg   = (const float*)d_in[2];  // [E, H, I]
    const float* wu   = (const float*)d_in[3];  // [E, H, I]
    const float* wd   = (const float*)d_in[4];  // [E, I, H]
    const float* wsgu = (const float*)d_in[5];  // [H, 2*Is]
    const float* wsd  = (const float*)d_in[6];  // [Is, H]
    float* out = (float*)d_out;

    char* p = (char*)d_ws;
    auto take = [&](size_t bytes) {
        char* q = p;
        p += (bytes + 255) & ~(size_t)255;
        return q;
    };
    __bf16* xb    = (__bf16*)take((size_t)T_TOK * HID * 2);
    __bf16* wgt   = (__bf16*)take((size_t)NEXP * HID * MOE_I * 2);  // [E][I][H]
    __bf16* wut   = (__bf16*)take((size_t)NEXP * HID * MOE_I * 2);  // [E][I][H]
    __bf16* wdt   = (__bf16*)take((size_t)NEXP * MOE_I * HID * 2);  // [E][H][I]
    __bf16* wsgut = (__bf16*)take((size_t)HID * SH_GU * 2);         // [2048][H]
    __bf16* wsdt  = (__bf16*)take((size_t)SH_I * HID * 2);          // [H][Is]
    __bf16* hb    = (__bf16*)take((size_t)NEXP * T_TOK * MOE_I * 2);
    __bf16* hsb   = (__bf16*)take((size_t)T_TOK * SH_I * 2);
    float*  combine = (float*)take((size_t)T_TOK * NEXP * 4);

    // 1) convert (+ transpose weights so K is the contiguous dim everywhere)
    cvt_f32_bf16<<<2048, 256, 0, stream>>>(x, xb, T_TOK * HID);
    {
        dim3 g(MOE_I / 32, HID / 32, NEXP);   // [H][I] -> [I][H]
        cvt_transpose_bf16<<<g, 256, 0, stream>>>(wg, wgt, HID, MOE_I);
        cvt_transpose_bf16<<<g, 256, 0, stream>>>(wu, wut, HID, MOE_I);
    }
    {
        dim3 g(HID / 32, MOE_I / 32, NEXP);   // [I][H] -> [H][I]
        cvt_transpose_bf16<<<g, 256, 0, stream>>>(wd, wdt, MOE_I, HID);
    }
    {
        dim3 g(SH_GU / 32, HID / 32, 1);      // [H][2048] -> [2048][H]
        cvt_transpose_bf16<<<g, 256, 0, stream>>>(wsgu, wsgut, HID, SH_GU);
    }
    {
        dim3 g(HID / 32, SH_I / 32, 1);       // [Is][H] -> [H][Is]
        cvt_transpose_bf16<<<g, 256, 0, stream>>>(wsd, wsdt, SH_I, HID);
    }

    // 2) router -> dense combine matrix
    router_kernel<<<T_TOK / 256, 256, 0, stream>>>(x, rw, combine);

    // 3) routed experts: fused gate/up + SwiGLU -> hb[e][t][i]
    dim3 g_ru(MOE_I / 64, T_TOK / 128, NEXP);
    gateup_kernel<<<g_ru, 256, 0, stream>>>(
        xb, wgt, wut, (long)HID * MOE_I,
        hb, (long)T_TOK * MOE_I, MOE_I, HID);

    // 4) shared expert: gate = rows [0,1024) of wsgut, up = rows [1024,2048)
    dim3 g_sh(SH_I / 64, T_TOK / 128, 1);
    gateup_kernel<<<g_sh, 256, 0, stream>>>(
        xb, wsgut, wsgut + (size_t)SH_I * HID, 0L,
        hsb, 0L, SH_I, HID);

    // 5) routed down-proj with combine weighting (writes out)
    dim3 g_dn(HID / 64, T_TOK / 128, 1);
    down_routed_kernel<<<g_dn, 256, 0, stream>>>(hb, wdt, combine, out);

    // 6) shared down-proj (adds to out)
    down_shared_kernel<<<g_dn, 256, 0, stream>>>(hsb, wsdt, out);
}
